// Conv1D_style_74191265071761
// MI455X (gfx1250) — matrix operand order repaired and checked by reading, not compiled, not run on hardware
//
#include <hip/hip_runtime.h>
#include <stdint.h>

// Problem constants (match reference)
#define B_ROWS 8192
#define DIN    4096
#define DOUT   4096
#define NC     64

// GEMM tiling
#define BM  128
#define BN  128
#define BK  32
#define LDA 40   // LDS row stride in halves (80B = 5x16B: keeps b128 alignment, skews banks)
#define LDB 40
#define KTILES (DIN / BK)
#define ATILE_BYTES (BM * LDA * 2)   // 10240
#define BTILE_BYTES (BN * LDB * 2)   // 10240

// prep (K=64) tiling
#define PLD 72   // LDS row stride in halves for 64-wide tiles (144B = 9x16B)

typedef __attribute__((ext_vector_type(16))) __bf16 v16bf;
typedef __attribute__((ext_vector_type(8)))  float  v8f;

union Frag {
    v16bf v;
    uint4 q[2];
};

__device__ __forceinline__ uint32_t f32_to_bf16_bits(float f) {
    uint32_t u = __float_as_uint(f);
    return (u + 0x7FFFu + ((u >> 16) & 1u)) >> 16;   // round-to-nearest-even
}
__device__ __forceinline__ float bf16_bits_to_f32(uint32_t h) {
    return __uint_as_float(h << 16);
}

// CDNA5 async global->LDS copy (16B per lane), GVS mode: saddr(u64) + voff(i32).
// VDST carries the LDS byte address (low 32 bits of the flat shared pointer,
// per ISA 10.2). Tracked by ASYNCcnt.
__device__ __forceinline__ void async_load_b128(uint32_t lds_addr, uint32_t voff,
                                                uint64_t sbase) {
    asm volatile("global_load_async_to_lds_b128 %0, %1, %2"
                 :: "v"(lds_addr), "v"(voff), "s"(sbase)
                 : "memory");
}
__device__ __forceinline__ void wait_asynccnt0() {
    asm volatile("s_wait_asynccnt 0" ::: "memory");
}

// ---------------------------------------------------------------------------
// Elementwise f32 -> bf16 convert (used for cluster).
// ---------------------------------------------------------------------------
__global__ void __launch_bounds__(256) cvt_bf16(const float* __restrict__ src,
                                                uint16_t* __restrict__ dst, int n) {
    int i = blockIdx.x * 256 + threadIdx.x;
    const int stride = gridDim.x * 256;
    for (; i < n; i += stride) dst[i] = (uint16_t)f32_to_bf16_bits(src[i]);
}

// ---------------------------------------------------------------------------
// Transpose-convert f32 S[R][C] -> bf16 D[C][R] via 32x32 LDS tiles.
// Used for weight (DIN x DOUT -> DOUT x DIN) and styles (64 x 4096 -> 4096 x 64).
// Grid: (C/32, R/32), block (32,8).
// ---------------------------------------------------------------------------
__global__ void __launch_bounds__(256) transpose_bf16(const float* __restrict__ S,
                                                      uint16_t* __restrict__ D,
                                                      int R, int C) {
    __shared__ float tile[32][33];
    const int r0 = blockIdx.y * 32;
    const int c0 = blockIdx.x * 32;
    const int tx = threadIdx.x;
    const int ty = threadIdx.y;
#pragma unroll
    for (int k = 0; k < 4; ++k) {
        const int r = ty + k * 8;
        tile[r][tx] = S[(size_t)(r0 + r) * C + c0 + tx];
    }
    __syncthreads();
#pragma unroll
    for (int k = 0; k < 4; ++k) {
        const int r = ty + k * 8;
        D[(size_t)(c0 + r) * R + r0 + tx] = (uint16_t)f32_to_bf16_bits(tile[tx][r]);
    }
}

// ---------------------------------------------------------------------------
// WMMA prep: tmpL = cluster@style_L, tmpR = cluster@style_R (K=64, 2 k-steps),
// fused epilogue: Aw = bf16(x * tmpL), Rw = bf16(tmpR).
// 128x128 block, 8 waves (2x4), wave tile 64x32. Two phases (L then R) so only
// one 64-VGPR accumulator set is live at a time.
// ---------------------------------------------------------------------------
__global__ void __launch_bounds__(256) prep_wmma(const float* __restrict__ x,
                                                 const uint16_t* __restrict__ Cb,   // [B_ROWS][64]
                                                 const uint16_t* __restrict__ SLt,  // [DIN][64]
                                                 const uint16_t* __restrict__ SRt,  // [DOUT][64]
                                                 uint16_t* __restrict__ Aw,
                                                 uint16_t* __restrict__ Rw) {
    __shared__ uint16_t Cs[BM * PLD];
    __shared__ uint16_t Ls[BN * PLD];
    __shared__ uint16_t Rs[BN * PLD];

    const int tid  = threadIdx.x;
    const int lane = tid & 31;
    const int wave = tid >> 5;
    const int wm   = wave & 1;
    const int wn   = wave >> 1;
    const int rb   = blockIdx.y * BM;
    const int cb   = blockIdx.x * BN;
    const int ml   = lane & 15;
    const int hi   = lane >> 4;

    const uint64_t bC = (uint64_t)(uintptr_t)Cb;
    const uint64_t bL = (uint64_t)(uintptr_t)SLt;
    const uint64_t bR = (uint64_t)(uintptr_t)SRt;
    const uint32_t lC = (uint32_t)(uintptr_t)&Cs[0];
    const uint32_t lL = (uint32_t)(uintptr_t)&Ls[0];
    const uint32_t lR = (uint32_t)(uintptr_t)&Rs[0];

    // Each tile: 128 rows x 64 halves = 8 x 16B chunks per row -> 1024 chunks.
#pragma unroll
    for (int j = 0; j < 4; ++j) {
        const int id  = tid + j * 256;
        const int row = id >> 3;
        const int cg  = id & 7;
        const uint32_t lofs  = (uint32_t)((row * PLD + cg * 8) * 2);
        const uint32_t voffM = (uint32_t)(((rb + row) * NC + cg * 8) * 2);
        const uint32_t voffN = (uint32_t)(((cb + row) * NC + cg * 8) * 2);
        async_load_b128(lC + lofs, voffM, bC);
        async_load_b128(lL + lofs, voffN, bL);
        async_load_b128(lR + lofs, voffN, bR);
    }
    wait_asynccnt0();
    __syncthreads();

    auto phase = [&](const uint16_t* stile, uint16_t* dst, bool mulx) {
        v8f acc[4][2];
#pragma unroll
        for (int mi = 0; mi < 4; ++mi)
#pragma unroll
            for (int ni = 0; ni < 2; ++ni)
                acc[mi][ni] = {};
#pragma unroll
        for (int s = 0; s < 2; ++s) {
            const int kbA = s * 32 + hi * 8;
            const int kbB = s * 32 + hi * 8;
            Frag afr[4], bfr[2];
#pragma unroll
            for (int mi = 0; mi < 4; ++mi) {
                const uint16_t* p = Cs + (wm * 64 + mi * 16 + ml) * PLD;
                afr[mi].q[0] = *(const uint4*)(p + kbA);
                afr[mi].q[1] = *(const uint4*)(p + kbA + 16);
            }
#pragma unroll
            for (int ni = 0; ni < 2; ++ni) {
                const uint16_t* p = stile + (wn * 32 + ni * 16 + ml) * PLD + kbB;
                bfr[ni].q[0] = *(const uint4*)(p);
                bfr[ni].q[1] = *(const uint4*)(p + 16);
            }
#pragma unroll
            for (int mi = 0; mi < 4; ++mi)
#pragma unroll
                for (int ni = 0; ni < 2; ++ni)
                    acc[mi][ni] = __builtin_amdgcn_wmma_f32_16x16x32_bf16(
                        false, afr[mi].v, false, bfr[ni].v,
                        (short)0, acc[mi][ni], false, false);
        }
        // Epilogue: C layout (lane<16: M=r; lane>=16: M=8+r; N=lane&15)
#pragma unroll
        for (int mi = 0; mi < 4; ++mi) {
#pragma unroll
            for (int ni = 0; ni < 2; ++ni) {
                const int col   = cb + wn * 32 + ni * 16 + ml;
                const int rbase = rb + wm * 64 + mi * 16 + hi * 8;
#pragma unroll
                for (int r = 0; r < 8; ++r) {
                    const int row = rbase + r;
                    float v = acc[mi][ni][r];
                    if (mulx) v *= x[(size_t)row * DIN + col];
                    dst[(size_t)row * DIN + col] = (uint16_t)f32_to_bf16_bits(v);
                }
            }
        }
    };
    phase(Ls, Aw, true);    // Aw = bf16(x * (cluster@style_L))
    phase(Rs, Rw, false);   // Rw = bf16(cluster@style_R)
}

// ---------------------------------------------------------------------------
// Main GEMM: out[b,o] = (sum_k Aw[b,k]*Wt[o,k]) * Rw[b,o]
// 128x128 block tile, BK=32, 8 waves (2x4), wave tile 64x32 = 4x2 WMMA tiles.
// Double-buffered LDS fed by GLOBAL_LOAD_ASYNC_TO_LDS_B128 (ASYNCcnt).
// K-loop forced to unroll(1) + runtime buffer offset so each accumulator
// stays pinned in one 8-VGPR tuple (no copy storm).
// ---------------------------------------------------------------------------
__global__ void __launch_bounds__(256) gemm_mod(const uint16_t* __restrict__ Aw,
                                                const uint16_t* __restrict__ Wt,
                                                const uint16_t* __restrict__ Rw,
                                                float* __restrict__ out) {
    __shared__ uint16_t As[2][BM * LDA];
    __shared__ uint16_t Bs[2][BN * LDB];

    const int tid  = threadIdx.x;
    const int lane = tid & 31;
    const int wave = tid >> 5;
    const int wm   = wave & 1;   // M offset wm*64
    const int wn   = wave >> 1;  // N offset wn*32
    const int rb   = blockIdx.y * BM;
    const int cb   = blockIdx.x * BN;

    const uint64_t baseA = (uint64_t)(uintptr_t)Aw;
    const uint64_t baseB = (uint64_t)(uintptr_t)Wt;

    // Two 16B chunks per thread per tile: id = tid, tid+256 -> (row, chunk).
    uint32_t voffA[2], voffB[2];      // global byte offsets at k=0
    uint32_t ldsA0[2], ldsB0[2];      // LDS byte addresses in buffer 0
#pragma unroll
    for (int j = 0; j < 2; ++j) {
        const int id  = tid + j * 256;
        const int row = id >> 2;      // 0..127
        const int cg  = id & 3;       // 16B chunk in 32-half row
        voffA[j] = (uint32_t)(((rb + row) * DIN + cg * 8) * 2);
        voffB[j] = (uint32_t)(((cb + row) * DIN + cg * 8) * 2);
        ldsA0[j] = (uint32_t)(uintptr_t)&As[0][row * LDA + cg * 8];
        ldsB0[j] = (uint32_t)(uintptr_t)&Bs[0][row * LDB + cg * 8];
    }

    auto issue_async = [&](int kt, int buf) {
        const uint32_t ko  = (uint32_t)(kt * BK * 2);        // byte advance along K
        const uint32_t boA = (uint32_t)(buf * ATILE_BYTES);  // buffer select
        const uint32_t boB = (uint32_t)(buf * BTILE_BYTES);
#pragma unroll
        for (int j = 0; j < 2; ++j) {
            async_load_b128(ldsA0[j] + boA, voffA[j] + ko, baseA);
            async_load_b128(ldsB0[j] + boB, voffB[j] + ko, baseB);
        }
    };

    v8f acc[4][2];
#pragma unroll
    for (int mi = 0; mi < 4; ++mi)
#pragma unroll
        for (int ni = 0; ni < 2; ++ni)
            acc[mi][ni] = {};

    const int ml  = lane & 15;
    const int hi  = lane >> 4;
    const int kbA = hi * 8;    // A frag: lanes<16 K0..7 & K16..23; >=16 K8..15 & K24..31
    const int kbB = hi * 8;    // B frag: lanes<16 K0..7 & K16..23; >=16 K8..15 & K24..31

    auto compute = [&](int buf) {
        const uint16_t* as = (const uint16_t*)((const char*)&As[0][0] + buf * ATILE_BYTES);
        const uint16_t* bs = (const uint16_t*)((const char*)&Bs[0][0] + buf * BTILE_BYTES);
        Frag afr[4], bfr[2];
#pragma unroll
        for (int mi = 0; mi < 4; ++mi) {
            const uint16_t* p = as + (wm * 64 + mi * 16 + ml) * LDA;
            afr[mi].q[0] = *(const uint4*)(p + kbA);
            afr[mi].q[1] = *(const uint4*)(p + kbA + 16);
        }
#pragma unroll
        for (int ni = 0; ni < 2; ++ni) {
            const uint16_t* p = bs + (wn * 32 + ni * 16 + ml) * LDB + kbB;
            bfr[ni].q[0] = *(const uint4*)(p);
            bfr[ni].q[1] = *(const uint4*)(p + 16);
        }
#pragma unroll
        for (int mi = 0; mi < 4; ++mi)
#pragma unroll
            for (int ni = 0; ni < 2; ++ni)
                acc[mi][ni] = __builtin_amdgcn_wmma_f32_16x16x32_bf16(
                    false, afr[mi].v, false, bfr[ni].v,
                    (short)0, acc[mi][ni], false, false);
    };

    issue_async(0, 0);

#pragma unroll 1
    for (int kt = 0; kt < KTILES; ++kt) {
        wait_asynccnt0();          // tile kt's DMA (issued last iteration) done
        __syncthreads();           // visible to all waves; all reads of buf^1 done
        if (kt + 1 < KTILES) issue_async(kt + 1, (kt + 1) & 1);
        compute(kt & 1);
    }

    // Epilogue: C layout per ISA (lane<16: M=r; lane>=16: M=8+r; N = lane&15)
#pragma unroll
    for (int mi = 0; mi < 4; ++mi) {
#pragma unroll
        for (int ni = 0; ni < 2; ++ni) {
            const int col   = cb + wn * 32 + ni * 16 + (lane & 15);
            const int rbase = rb + wm * 64 + mi * 16 + ((lane >> 4) << 3);
#pragma unroll
            for (int r = 0; r < 8; ++r) {
                const int row = rbase + r;
                const float s = bf16_bits_to_f32(Rw[(size_t)row * DOUT + col]);
                out[(size_t)row * DOUT + col] = acc[mi][ni][r] * s;
            }
        }
    }
}

// ---------------------------------------------------------------------------
// Launcher. d_in order: x, cluster, weight, style_L, style_R.
// Workspace: Aw (64MB) | Wt (32MB) | Rw (64MB) | Cb (1MB) | SLt (.5MB) | SRt (.5MB)
// ---------------------------------------------------------------------------
extern "C" void kernel_launch(void* const* d_in, const int* in_sizes, int n_in,
                              void* d_out, int out_size, void* d_ws, size_t ws_size,
                              hipStream_t stream) {
    const float* x       = (const float*)d_in[0];
    const float* cluster = (const float*)d_in[1];
    const float* weight  = (const float*)d_in[2];
    const float* style_L = (const float*)d_in[3];
    const float* style_R = (const float*)d_in[4];
    float* out = (float*)d_out;

    uint16_t* Aw  = (uint16_t*)d_ws;                    // B_ROWS*DIN
    uint16_t* Wt  = Aw  + (size_t)B_ROWS * DIN;         // DOUT*DIN (transposed weight)
    uint16_t* Rw  = Wt  + (size_t)DOUT * DIN;           // B_ROWS*DOUT
    uint16_t* Cb  = Rw  + (size_t)B_ROWS * DOUT;        // B_ROWS*NC
    uint16_t* SLt = Cb  + (size_t)B_ROWS * NC;          // DIN*NC (transposed style_L)
    uint16_t* SRt = SLt + (size_t)DIN * NC;             // DOUT*NC (transposed style_R)

    cvt_bf16<<<512, 256, 0, stream>>>(cluster, Cb, B_ROWS * NC);
    transpose_bf16<<<dim3(DIN / 32, NC / 32), dim3(32, 8), 0, stream>>>(style_L, SLt, NC, DIN);
    transpose_bf16<<<dim3(DOUT / 32, NC / 32), dim3(32, 8), 0, stream>>>(style_R, SRt, NC, DOUT);
    transpose_bf16<<<dim3(DOUT / 32, DIN / 32), dim3(32, 8), 0, stream>>>(weight, Wt, DIN, DOUT);
    prep_wmma<<<dim3(DOUT / BN, B_ROWS / BM), 256, 0, stream>>>(x, Cb, SLt, SRt, Aw, Rw);
    gemm_mod<<<dim3(DOUT / BN, B_ROWS / BM), 256, 0, stream>>>(Aw, Wt, Rw, out);
}
